// EGraphConditionEncoder_14250701488265
// MI455X (gfx1250) — matrix operand order repaired and checked by reading, not compiled, not run on hardware
//
#include <hip/hip_runtime.h>
#include <hip/hip_bf16.h>

#define N_NODES 50000
#define N_EDGES 600000
#define N_GRAPHS 64
#define IN_DIM 16
#define HID 128
#define OUT_DIM 256
#define N_LAYERS 3
#define WPB 4   // waves per block (wave32)

typedef __attribute__((ext_vector_type(16))) __bf16 v16bf;
typedef __attribute__((ext_vector_type(8)))  __bf16 v8bf;
typedef __attribute__((ext_vector_type(8)))  float  v8f;

__device__ __forceinline__ float silu_f(float v) {
    return v * (1.0f / (1.0f + __expf(-v)));
}

// A fragment: 16x32 bf16 tile, row M = lane%16 from LDS (row-major, stride ld elements).
// Per-lane data = 16 contiguous elems: k0+hi*16+[0..15]
// (same k order as load_b_frag) -> two ds_load_b128, no packing movs.
__device__ __forceinline__ v16bf load_a_frag(const __bf16* S, int ld, int k0) {
    const int lane = threadIdx.x & 31;
    const int m  = lane & 15;
    const int hi = (lane >> 4) & 1;
    const __bf16* base = S + m * ld + k0 + hi * 16;
    v8bf lo = *(const v8bf*)(base);
    v8bf hi8 = *(const v8bf*)(base + 8);
    return __builtin_shufflevector(lo, hi8, 0, 1, 2, 3, 4, 5, 6, 7,
                                   8, 9, 10, 11, 12, 13, 14, 15);
}

// B fragment: 32x16 tile, col N = n0 + lane%16, from TRANSPOSED weights Wt[N][K]
// (row-major, K stride). Per-lane data = 16 contiguous bf16 at k0+hi*16
// -> one 32-byte load (two global_load_b128), no packing movs.
__device__ __forceinline__ v16bf load_b_frag(const __bf16* Wt, int Kld, int k0, int n0) {
    const int lane = threadIdx.x & 31;
    const int n  = n0 + (lane & 15);
    const int hi = (lane >> 4) & 1;
    return *(const v16bf*)(Wt + (size_t)n * Kld + k0 + hi * 16);
}

__device__ __forceinline__ v8f wmma_bf16(v16bf a, v16bf b, v8f c) {
    return __builtin_amdgcn_wmma_f32_16x16x32_bf16(false, a, false, b, (short)0, c, false, false);
}

// ---------------- small utility kernels ----------------

__global__ void zero_f32_k(float* __restrict__ p, int n) {
    int i = blockIdx.x * blockDim.x + threadIdx.x;
    if (i < n) p[i] = 0.0f;
}

// transpose-convert: src row-major [K x N] (per layer, layer stride srcLS)
// -> dst row-major [N x K] bf16, layer stride N*K
__global__ void cvtT_k(const float* __restrict__ s, __bf16* __restrict__ d,
                       int K, int N, int srcLS, int total) {
    int idx = blockIdx.x * blockDim.x + threadIdx.x;
    if (idx >= total) return;
    int perLayer = N * K;
    int l = idx / perLayer;
    int rem = idx - l * perLayer;
    int n = rem / K;
    int k = rem - n * K;
    d[idx] = (__bf16)s[(size_t)l * srcLS + (size_t)k * N + n];
}

__global__ void deg_k(const int* __restrict__ row, float* __restrict__ deg, int e) {
    int i = blockIdx.x * blockDim.x + threadIdx.x;
    if (i < e) atomicAdd(&deg[row[i]], 1.0f);
}

__global__ void gcnt_k(const int* __restrict__ batch, float* __restrict__ gcnt, int n) {
    int i = blockIdx.x * blockDim.x + threadIdx.x;
    if (i < n) atomicAdd(&gcnt[batch[i]], 1.0f);
}

// h = x @ emb_in_w + b   (K=16, tiny)
__global__ void embed_k(const float* __restrict__ x, const float* __restrict__ w,
                        const float* __restrict__ b, float* __restrict__ h, int n) {
    int idx = blockIdx.x * blockDim.x + threadIdx.x;
    if (idx >= n * HID) return;
    int nd = idx >> 7, c = idx & 127;
    float s = b[c];
#pragma unroll
    for (int k = 0; k < IN_DIM; ++k) s += x[nd * IN_DIM + k] * w[k * HID + c];
    h[idx] = s;
}

// coord_next = coord + coordAcc / max(deg,1)
__global__ void coord_upd_k(const float* __restrict__ c, const float* __restrict__ acc,
                            const float* __restrict__ deg, float* __restrict__ out, int n) {
    int idx = blockIdx.x * blockDim.x + threadIdx.x;
    if (idx >= n * 3) return;
    int nd = idx / 3;
    float d = deg[nd]; d = d < 1.0f ? 1.0f : d;
    out[idx] = c[idx] + acc[idx] / d;
}

// h_out = h @ emb_out_w + b, then atomic segment-sum into gsum[batch]
__global__ void embout_pool_k(const float* __restrict__ h, const float* __restrict__ w,
                              const float* __restrict__ b, const int* __restrict__ batch,
                              float* __restrict__ gsum, int n) {
    int idx = blockIdx.x * blockDim.x + threadIdx.x;
    if (idx >= n * HID) return;
    int nd = idx >> 7, c = idx & 127;
    float s = b[c];
#pragma unroll 8
    for (int k = 0; k < HID; ++k) s += h[nd * HID + k] * w[k * HID + c];
    atomicAdd(&gsum[batch[nd] * HID + c], s);
}

// out = (gsum/cnt) @ fc_w + fc_b
__global__ void fc_k(const float* __restrict__ gsum, const float* __restrict__ gcnt,
                     const float* __restrict__ w, const float* __restrict__ b,
                     float* __restrict__ out) {
    int idx = blockIdx.x * blockDim.x + threadIdx.x;
    if (idx >= N_GRAPHS * OUT_DIM) return;
    int g = idx / OUT_DIM, o = idx % OUT_DIM;
    float cnt = gcnt[g]; cnt = cnt < 1.0f ? 1.0f : cnt;
    float inv = 1.0f / cnt;
    float s = b[o];
#pragma unroll 8
    for (int k = 0; k < HID; ++k) s += gsum[g * HID + k] * inv * w[k * OUT_DIM + o];
    out[idx] = s;
}

// ---------------- fused edge kernel ----------------
// Per wave: 16 edges. m1 = silu([h_row|h_col|rad|ea]@W1+b1); m = silu(m1@W2+b2);
// cw = (silu(m@CW1+cb1))@cw2;  scatter m -> magg[row], diff*cw -> coordAcc[row].
__global__ __launch_bounds__(32 * WPB, 1) void edge_kernel(
    const float* __restrict__ h, const float* __restrict__ coord,
    const float* __restrict__ eattr,
    const int* __restrict__ row, const int* __restrict__ col,
    const __bf16* __restrict__ w1t,   // transposed [128][256] per layer
    const float* __restrict__ w1f,    // original f32 [258][128] (tail rows 256,257)
    const float* __restrict__ b1,
    const __bf16* __restrict__ w2t,   // transposed [128][128]
    const float* __restrict__ b2,
    const __bf16* __restrict__ cw1t,  // transposed [128][128]
    const float* __restrict__ cb1,
    const float* __restrict__ cw2,
    float* __restrict__ magg, float* __restrict__ coordAcc, int E)
{
    __shared__ __bf16 hrowS[WPB][16][HID];
    __shared__ __bf16 hcolS[WPB][16][HID];
    __shared__ __bf16 mS[WPB][16][HID];
    __shared__ float  diffS[WPB][16][3];
    __shared__ float  radS[WPB][16];
    __shared__ float  eaS[WPB][16];
    __shared__ float  cwS[WPB][16];
    __shared__ int    rowS[WPB][16];

    const int wave = threadIdx.x >> 5;
    const int lane = threadIdx.x & 31;
    const int e0 = (blockIdx.x * WPB + wave) * 16;

    __builtin_prefetch(w1t, 0, 3);
    __builtin_prefetch(w2t, 0, 3);
    __builtin_prefetch(cw1t, 0, 3);

    // ---- stage 16 edges: gathered h rows (coalesced float4), geometry ----
    for (int r = 0; r < 16; ++r) {
        int e = e0 + r; if (e >= E) e = E - 1;
        int ri = row[e], ci = col[e];
        float4 vr = ((const float4*)(h + (size_t)ri * HID))[lane];
        float4 vc = ((const float4*)(h + (size_t)ci * HID))[lane];
        __bf16* dr = &hrowS[wave][r][lane * 4];
        __bf16* dc = &hcolS[wave][r][lane * 4];
        dr[0] = (__bf16)vr.x; dr[1] = (__bf16)vr.y; dr[2] = (__bf16)vr.z; dr[3] = (__bf16)vr.w;
        dc[0] = (__bf16)vc.x; dc[1] = (__bf16)vc.y; dc[2] = (__bf16)vc.z; dc[3] = (__bf16)vc.w;
    }
    if (lane < 16) {
        int e = e0 + lane; if (e >= E) e = E - 1;
        int ri = row[e], ci = col[e];
        rowS[wave][lane] = ri;
        float rad = 0.0f;
#pragma unroll
        for (int k = 0; k < 3; ++k) {
            float d = coord[ri * 3 + k] - coord[ci * 3 + k];
            diffS[wave][lane][k] = d;
            rad += d * d;
        }
        radS[wave][lane] = rad;
        eaS[wave][lane] = eattr[e];
    }
    __syncthreads();

    const int hi = lane >> 4;
    const int nb = lane & 15;
    const v8f vz = {0.f, 0.f, 0.f, 0.f, 0.f, 0.f, 0.f, 0.f};

    // ---- GEMM 1: [16 x 258] @ [258 x 128] as two K=128 GEMMs + rank-2 tail ----
    v8f acc[8];
#pragma unroll
    for (int t = 0; t < 8; ++t) acc[t] = vz;
#pragma unroll
    for (int kc = 0; kc < 4; ++kc) {
        v16bf a = load_a_frag(&hrowS[wave][0][0], HID, kc * 32);
#pragma unroll
        for (int t = 0; t < 8; ++t)
            acc[t] = wmma_bf16(a, load_b_frag(w1t, 256, kc * 32, t * 16), acc[t]);
    }
#pragma unroll
    for (int kc = 0; kc < 4; ++kc) {
        v16bf a = load_a_frag(&hcolS[wave][0][0], HID, kc * 32);
#pragma unroll
        for (int t = 0; t < 8; ++t)
            acc[t] = wmma_bf16(a, load_b_frag(w1t, 256, HID + kc * 32, t * 16), acc[t]);
    }
#pragma unroll
    for (int t = 0; t < 8; ++t) {
        int n = t * 16 + nb;
        float wr = w1f[256 * HID + n];
        float we = w1f[257 * HID + n];
        float bb = b1[n];
#pragma unroll
        for (int r = 0; r < 8; ++r) {
            int m = r + hi * 8;
            float v = acc[t][r] + bb + radS[wave][m] * wr + eaS[wave][m] * we;
            mS[wave][m][n] = (__bf16)silu_f(v);
        }
    }
    __syncthreads();

    // ---- GEMM 2: m = silu(m1 @ W2 + b2); scatter into magg ----
#pragma unroll
    for (int t = 0; t < 8; ++t) acc[t] = vz;
#pragma unroll
    for (int kc = 0; kc < 4; ++kc) {
        v16bf a = load_a_frag(&mS[wave][0][0], HID, kc * 32);
#pragma unroll
        for (int t = 0; t < 8; ++t)
            acc[t] = wmma_bf16(a, load_b_frag(w2t, HID, kc * 32, t * 16), acc[t]);
    }
#pragma unroll
    for (int t = 0; t < 8; ++t) {
        int n = t * 16 + nb;
        float bb = b2[n];
#pragma unroll
        for (int r = 0; r < 8; ++r) {
            int m = r + hi * 8;
            float v = silu_f(acc[t][r] + bb);
            mS[wave][m][n] = (__bf16)v;
            int e = e0 + m;
            if (e < E) atomicAdd(&magg[(size_t)rowS[wave][m] * HID + n], v);
        }
    }
    __syncthreads();

    // ---- GEMM 3: p = silu(m @ CW1 + cb1), cw = p . cw2 ----
#pragma unroll
    for (int t = 0; t < 8; ++t) acc[t] = vz;
#pragma unroll
    for (int kc = 0; kc < 4; ++kc) {
        v16bf a = load_a_frag(&mS[wave][0][0], HID, kc * 32);
#pragma unroll
        for (int t = 0; t < 8; ++t)
            acc[t] = wmma_bf16(a, load_b_frag(cw1t, HID, kc * 32, t * 16), acc[t]);
    }
    float part[8];
#pragma unroll
    for (int r = 0; r < 8; ++r) part[r] = 0.0f;
#pragma unroll
    for (int t = 0; t < 8; ++t) {
        int n = t * 16 + nb;
        float cb = cb1[n];
        float w = cw2[n];
#pragma unroll
        for (int r = 0; r < 8; ++r)
            part[r] += silu_f(acc[t][r] + cb) * w;
    }
    // reduce across the 16 lanes of each half (lanes share the same 8 rows)
#pragma unroll
    for (int off = 1; off < 16; off <<= 1)
#pragma unroll
        for (int r = 0; r < 8; ++r)
            part[r] += __shfl_xor(part[r], off, 32);
    if (nb == 0)
#pragma unroll
        for (int r = 0; r < 8; ++r)
            cwS[wave][hi * 8 + r] = part[r];
    __syncthreads();

    if (lane < 16) {
        int e = e0 + lane;
        if (e < E) {
            float cwv = cwS[wave][lane];
            int ri = rowS[wave][lane];
#pragma unroll
            for (int k = 0; k < 3; ++k)
                atomicAdd(&coordAcc[(size_t)ri * 3 + k], diffS[wave][lane][k] * cwv);
        }
    }
}

// ---------------- fused node kernel ----------------
// h_out = h + silu([h|magg] @ W1 + b1) @ W2 + b2
__global__ __launch_bounds__(32 * WPB, 1) void node_kernel(
    const float* __restrict__ h, const float* __restrict__ magg,
    const __bf16* __restrict__ w1t,   // transposed [128][256]
    const float* __restrict__ b1,
    const __bf16* __restrict__ w2t,   // transposed [128][128]
    const float* __restrict__ b2,
    float* __restrict__ hout, int N)
{
    __shared__ __bf16 nfS[WPB][16][2 * HID];
    __shared__ __bf16 qS[WPB][16][HID];

    const int wave = threadIdx.x >> 5;
    const int lane = threadIdx.x & 31;
    const int n0 = (blockIdx.x * WPB + wave) * 16;

    __builtin_prefetch(w1t, 0, 3);
    __builtin_prefetch(w2t, 0, 3);

    for (int r = 0; r < 16; ++r) {
        int nd = n0 + r; if (nd >= N) nd = N - 1;
        float4 vh = ((const float4*)(h + (size_t)nd * HID))[lane];
        float4 vm = ((const float4*)(magg + (size_t)nd * HID))[lane];
        __bf16* dh = &nfS[wave][r][lane * 4];
        __bf16* dm = &nfS[wave][r][HID + lane * 4];
        dh[0] = (__bf16)vh.x; dh[1] = (__bf16)vh.y; dh[2] = (__bf16)vh.z; dh[3] = (__bf16)vh.w;
        dm[0] = (__bf16)vm.x; dm[1] = (__bf16)vm.y; dm[2] = (__bf16)vm.z; dm[3] = (__bf16)vm.w;
    }
    __syncthreads();

    const int hi = lane >> 4;
    const int nb = lane & 15;
    const v8f vz = {0.f, 0.f, 0.f, 0.f, 0.f, 0.f, 0.f, 0.f};

    v8f acc[8];
#pragma unroll
    for (int t = 0; t < 8; ++t) acc[t] = vz;
#pragma unroll
    for (int kc = 0; kc < 8; ++kc) {
        v16bf a = load_a_frag(&nfS[wave][0][0], 2 * HID, kc * 32);
#pragma unroll
        for (int t = 0; t < 8; ++t)
            acc[t] = wmma_bf16(a, load_b_frag(w1t, 256, kc * 32, t * 16), acc[t]);
    }
#pragma unroll
    for (int t = 0; t < 8; ++t) {
        int n = t * 16 + nb;
        float bb = b1[n];
#pragma unroll
        for (int r = 0; r < 8; ++r) {
            int m = r + hi * 8;
            qS[wave][m][n] = (__bf16)silu_f(acc[t][r] + bb);
        }
    }
    __syncthreads();

#pragma unroll
    for (int t = 0; t < 8; ++t) acc[t] = vz;
#pragma unroll
    for (int kc = 0; kc < 4; ++kc) {
        v16bf a = load_a_frag(&qS[wave][0][0], HID, kc * 32);
#pragma unroll
        for (int t = 0; t < 8; ++t)
            acc[t] = wmma_bf16(a, load_b_frag(w2t, HID, kc * 32, t * 16), acc[t]);
    }
#pragma unroll
    for (int t = 0; t < 8; ++t) {
        int n = t * 16 + nb;
        float bb = b2[n];
#pragma unroll
        for (int r = 0; r < 8; ++r) {
            int m = r + hi * 8;
            int nd = n0 + m;
            if (nd < N)
                hout[(size_t)nd * HID + n] = h[(size_t)nd * HID + n] + acc[t][r] + bb;
        }
    }
}

// ---------------- host orchestration ----------------

static inline int cdiv_i(int a, int b) { return (a + b - 1) / b; }

extern "C" void kernel_launch(void* const* d_in, const int* in_sizes, int n_in,
                              void* d_out, int out_size, void* d_ws, size_t ws_size,
                              hipStream_t stream) {
    const float* x        = (const float*)d_in[0];
    const int*   eidx     = (const int*)d_in[1];
    const float* coord_in = (const float*)d_in[2];
    const float* eattr    = (const float*)d_in[3];
    const int*   batch    = (const int*)d_in[4];
    const float* emb_in_w = (const float*)d_in[5];
    const float* emb_in_b = (const float*)d_in[6];
    const float* edge_w1  = (const float*)d_in[7];
    const float* edge_b1  = (const float*)d_in[8];
    const float* edge_w2  = (const float*)d_in[9];
    const float* edge_b2  = (const float*)d_in[10];
    const float* node_w1  = (const float*)d_in[11];
    const float* node_b1  = (const float*)d_in[12];
    const float* node_w2  = (const float*)d_in[13];
    const float* node_b2  = (const float*)d_in[14];
    const float* coord_w1 = (const float*)d_in[15];
    const float* coord_b1 = (const float*)d_in[16];
    const float* coord_w2 = (const float*)d_in[17];
    const float* emb_out_w = (const float*)d_in[18];
    const float* emb_out_b = (const float*)d_in[19];
    const float* fc_w     = (const float*)d_in[20];
    const float* fc_b     = (const float*)d_in[21];

    const int* row = eidx;
    const int* col = eidx + N_EDGES;

    // workspace carve
    char* p = (char*)d_ws;
    auto carve = [&](size_t bytes) {
        char* r = p;
        p += (bytes + 255) & ~(size_t)255;
        return (void*)r;
    };
    float* hA       = (float*)carve((size_t)N_NODES * HID * 4);
    float* hB       = (float*)carve((size_t)N_NODES * HID * 4);
    float* magg     = (float*)carve((size_t)N_NODES * HID * 4);
    float* coordA   = (float*)carve((size_t)N_NODES * 3 * 4);
    float* coordB   = (float*)carve((size_t)N_NODES * 3 * 4);
    float* coordAcc = (float*)carve((size_t)N_NODES * 3 * 4);
    float* deg      = (float*)carve((size_t)N_NODES * 4);
    float* gsum     = (float*)carve((size_t)N_GRAPHS * HID * 4);
    float* gcnt     = (float*)carve((size_t)N_GRAPHS * 4);
    // transposed bf16 weights: Wt[N=128][K], K stride keeps 32B alignment
    __bf16* ew1t = (__bf16*)carve((size_t)N_LAYERS * HID * 256 * 2);
    __bf16* ew2t = (__bf16*)carve((size_t)N_LAYERS * HID * HID * 2);
    __bf16* nw1t = (__bf16*)carve((size_t)N_LAYERS * HID * 256 * 2);
    __bf16* nw2t = (__bf16*)carve((size_t)N_LAYERS * HID * HID * 2);
    __bf16* cw1t = (__bf16*)carve((size_t)N_LAYERS * HID * HID * 2);

    const int T = 256;

    // transpose-convert weights to bf16 (tiny; L2-resident; redone each call)
    cvtT_k<<<cdiv_i(N_LAYERS * HID * 256, T), T, 0, stream>>>(
        edge_w1, ew1t, 256, HID, 258 * HID, N_LAYERS * HID * 256);
    cvtT_k<<<cdiv_i(N_LAYERS * HID * HID, T), T, 0, stream>>>(
        edge_w2, ew2t, HID, HID, HID * HID, N_LAYERS * HID * HID);
    cvtT_k<<<cdiv_i(N_LAYERS * HID * 256, T), T, 0, stream>>>(
        node_w1, nw1t, 256, HID, 2 * HID * HID, N_LAYERS * HID * 256);
    cvtT_k<<<cdiv_i(N_LAYERS * HID * HID, T), T, 0, stream>>>(
        node_w2, nw2t, HID, HID, HID * HID, N_LAYERS * HID * HID);
    cvtT_k<<<cdiv_i(N_LAYERS * HID * HID, T), T, 0, stream>>>(
        coord_w1, cw1t, HID, HID, HID * HID, N_LAYERS * HID * HID);

    // degree
    zero_f32_k<<<cdiv_i(N_NODES, T), T, 0, stream>>>(deg, N_NODES);
    deg_k<<<cdiv_i(N_EDGES, T), T, 0, stream>>>(row, deg, N_EDGES);

    // input embedding
    embed_k<<<cdiv_i(N_NODES * HID, T), T, 0, stream>>>(x, emb_in_w, emb_in_b, hA, N_NODES);

    float* hcur = hA;
    float* hnxt = hB;
    const float* ccur = coord_in;
    float* cbuf[2] = {coordA, coordB};

    const int edge_blocks = cdiv_i(N_EDGES, 16 * WPB);
    const int node_blocks = cdiv_i(N_NODES, 16 * WPB);

    for (int l = 0; l < N_LAYERS; ++l) {
        zero_f32_k<<<cdiv_i(N_NODES * HID, T), T, 0, stream>>>(magg, N_NODES * HID);
        zero_f32_k<<<cdiv_i(N_NODES * 3, T), T, 0, stream>>>(coordAcc, N_NODES * 3);

        edge_kernel<<<edge_blocks, 32 * WPB, 0, stream>>>(
            hcur, ccur, eattr, row, col,
            ew1t + (size_t)l * HID * 256, edge_w1 + (size_t)l * 258 * HID,
            edge_b1 + (size_t)l * HID,
            ew2t + (size_t)l * HID * HID, edge_b2 + (size_t)l * HID,
            cw1t + (size_t)l * HID * HID, coord_b1 + (size_t)l * HID,
            coord_w2 + (size_t)l * HID,
            magg, coordAcc, N_EDGES);

        float* cnext = cbuf[l & 1];
        coord_upd_k<<<cdiv_i(N_NODES * 3, T), T, 0, stream>>>(ccur, coordAcc, deg, cnext, N_NODES);
        ccur = cnext;

        node_kernel<<<node_blocks, 32 * WPB, 0, stream>>>(
            hcur, magg,
            nw1t + (size_t)l * HID * 256, node_b1 + (size_t)l * HID,
            nw2t + (size_t)l * HID * HID, node_b2 + (size_t)l * HID,
            hnxt, N_NODES);
        float* tmp = hcur; hcur = hnxt; hnxt = tmp;
    }

    // output head: emb_out + global mean pool + fc
    zero_f32_k<<<cdiv_i(N_GRAPHS * HID, T), T, 0, stream>>>(gsum, N_GRAPHS * HID);
    zero_f32_k<<<cdiv_i(N_GRAPHS, T), T, 0, stream>>>(gcnt, N_GRAPHS);
    gcnt_k<<<cdiv_i(N_NODES, T), T, 0, stream>>>(batch, gcnt, N_NODES);
    embout_pool_k<<<cdiv_i(N_NODES * HID, T), T, 0, stream>>>(hcur, emb_out_w, emb_out_b, batch, gsum, N_NODES);
    fc_k<<<cdiv_i(N_GRAPHS * OUT_DIM, T), T, 0, stream>>>(gsum, gcnt, fc_w, fc_b, (float*)d_out);
}
